// SSMBlock_47863115547035
// MI455X (gfx1250) — matrix operand order repaired and checked by reading, not compiled, not run on hardware
//
#include <hip/hip_runtime.h>
#include <hip/hip_bf16.h>
#include <math.h>

// ---------------------------------------------------------------------------
// Types for CDNA5 WMMA (wave32)
// ---------------------------------------------------------------------------
typedef __attribute__((ext_vector_type(16))) _Float16 v16h;
typedef __attribute__((ext_vector_type(8)))  float    v8f;
typedef __attribute__((ext_vector_type(4)))  unsigned int u32x4;
typedef __attribute__((ext_vector_type(4)))  int      v4i;

union FragH { v16h v; u32x4 q[2]; };
union FragF { v8f  v; float f[8]; };

// ---------------------------------------------------------------------------
// CDNA5 async global->LDS copy (ASYNCcnt-tracked LDS DMA), with safe fallback
// Probe-revealed signature: (v4i AS1*, v4i AS3*, imm int, imm int)
// ---------------------------------------------------------------------------
#if defined(__has_builtin)
#  if __has_builtin(__builtin_amdgcn_global_load_async_to_lds_b128)
#    define HAVE_ASYNC_COPY 1
#  endif
#endif

__device__ __forceinline__ void copy16(_Float16* dst_lds, const _Float16* src_g)
{
#ifdef HAVE_ASYNC_COPY
    v4i* s_gen = (v4i*)(void*)const_cast<_Float16*>(src_g);
    v4i* d_gen = (v4i*)(void*)dst_lds;
    __builtin_amdgcn_global_load_async_to_lds_b128(
        (__attribute__((address_space(1))) v4i*)s_gen,
        (__attribute__((address_space(3))) v4i*)d_gen,
        0, 0);
#else
    *(u32x4*)dst_lds = *(const u32x4*)src_g;
#endif
}

__device__ __forceinline__ void wait_async()
{
#ifdef HAVE_ASYNC_COPY
#  if __has_builtin(__builtin_amdgcn_s_wait_asynccnt)
    __builtin_amdgcn_s_wait_asynccnt(0);
#  else
    asm volatile("s_wait_asynccnt 0x0" ::: "memory");
#  endif
#endif
}

// ---------------------------------------------------------------------------
// Problem constants (from reference)
// ---------------------------------------------------------------------------
constexpr int BATCH   = 16;
constexpr int SEQ     = 1024;
constexpr int DM      = 512;
constexpr int DSTATE  = 64;
constexpr int HEADDIM = 64;
constexpr int DIN     = 1024;                  // EXPAND*DM
constexpr int NH      = DIN / HEADDIM;         // 16
constexpr int CONVDIM = DIN + 2 * DSTATE;      // 1152
constexpr int DINPROJ = 2 * DIN + 2 * DSTATE + NH; // 2192
constexpr int MROWS   = BATCH * SEQ;           // 16384
constexpr int OUT_CH  = 1024;
constexpr int IN_CH   = SEQ;                   // 1024 (conv in-channels = seq dim)
constexpr int HCONV   = DM;                    // 512  (conv spatial dim = model dim)
constexpr float EPS_F = 1e-5f;

__device__ __forceinline__ float siluf(float x) { return x / (1.0f + __expf(-x)); }

// ---------------------------------------------------------------------------
// Elementwise f32 -> f16 packers
// ---------------------------------------------------------------------------
__global__ void cvt_f32_f16(const float* __restrict__ s, _Float16* __restrict__ d, int n) {
    int i = blockIdx.x * blockDim.x + threadIdx.x;
    if (i < n) d[i] = (_Float16)s[i];
}

// conv_w (OUT_CH, IN_CH, 3) -> packed [tap][oc][ic] f16 (each tap a row-major Bt)
__global__ void pack_convw(const float* __restrict__ w, _Float16* __restrict__ d, int n) {
    int i = blockIdx.x * blockDim.x + threadIdx.x;
    if (i >= n) return;
    const int t  = i / (OUT_CH * IN_CH);
    const int oc = (i / IN_CH) % OUT_CH;
    const int ic = i % IN_CH;
    d[i] = (_Float16)w[((size_t)oc * IN_CH + ic) * 3 + t];
}

// ---------------------------------------------------------------------------
// Generic WMMA GEMM:  C[M,N](f32) = A[M,K](f16) * Bt[N,K](f16)^T
// Block tile 128x64, 8 waves, each wave 2x2 of 16x16x32 WMMA tiles.
// Double-buffered LDS; global->LDS staging via async LDS-DMA when available.
// ---------------------------------------------------------------------------
__global__ __launch_bounds__(256)
void gemm_f16_wmma(const _Float16* __restrict__ A, const _Float16* __restrict__ Bt,
                   float* __restrict__ C, _Float16* __restrict__ Ch,
                   int M, int N, int K, int lda, int ldb, int ldc)
{
    __shared__ __attribute__((aligned(16))) _Float16 As[2][128 * 40];
    __shared__ __attribute__((aligned(16))) _Float16 Bs[2][64 * 40];

    const int tid  = threadIdx.x;
    const int m0   = blockIdx.y * 128;
    const int n0   = blockIdx.x * 64;
    const int lane = tid & 31;
    const int wave = tid >> 5;
    const int wm   = wave & 3;          // 0..3 -> which 32-row slice
    const int wn   = wave >> 2;         // 0..1 -> which 32-col slice
    const int lmod = lane & 15;
    const int lhi  = (lane >> 4) & 1;   // lanes 16-31 hold upper K-halves

    const int arow = tid >> 1, ahalf = tid & 1; // A tile: 128 rows x 2 x 8-half chunks at k 0 and 16
    const int brow = tid >> 2, bq    = tid & 3; // B tile: 64 rows  x 4 x  8-half chunks

    const bool afull = (m0 + 128 <= M);   // block-uniform
    const bool bfull = (n0 + 64 <= N);    // block-uniform

    auto stageA = [&](int k0, int buf) {
        _Float16* dst = &As[buf][arow * 40 + ahalf * 16];
        const _Float16* src = A + (size_t)(m0 + arow) * lda + k0 + ahalf * 16;
        if (afull) {
            copy16(dst, src);
        } else {
            u32x4 av = {0u, 0u, 0u, 0u};
            if (m0 + arow < M) av = *(const u32x4*)src;
            *(u32x4*)dst = av;
        }
    };
    auto stageB = [&](int k0, int buf) {
        _Float16* dst = &Bs[buf][brow * 40 + bq * 8];
        const _Float16* src = Bt + (size_t)(n0 + brow) * ldb + k0 + bq * 8;
        if (bfull) {
            copy16(dst, src);
        } else {
            u32x4 bv = {0u, 0u, 0u, 0u};
            if (n0 + brow < N) bv = *(const u32x4*)src;
            *(u32x4*)dst = bv;
        }
    };

    FragF acc[2][2];
    v8f zero = {};
#pragma unroll
    for (int i = 0; i < 2; ++i)
#pragma unroll
        for (int j = 0; j < 2; ++j) acc[i][j].v = zero;

    // Prologue: stage first K-slice
    stageA(0, 0);
    stageB(0, 0);
    wait_async();
    __syncthreads();

    int cur = 0;
    for (int k0 = 0; k0 < K; k0 += 32) {
        const int nxt = cur ^ 1;
        // Kick off async staging of the next slice while we compute this one
        if (k0 + 32 < K) {
            stageA(k0 + 32, nxt);
            stageB(k0 + 32, nxt);
        }

        FragH a[2], b[2];
#pragma unroll
        for (int i = 0; i < 2; ++i) {
            const int base = (wm * 32 + i * 16 + lmod) * 40 + lhi * 8;
            a[i].q[0] = *(const u32x4*)&As[cur][base];
            a[i].q[1] = *(const u32x4*)&As[cur][base + 16];
        }
#pragma unroll
        for (int j = 0; j < 2; ++j) {
            const int base = (wn * 32 + j * 16 + lmod) * 40 + lhi * 8;
            b[j].q[0] = *(const u32x4*)&Bs[cur][base];
            b[j].q[1] = *(const u32x4*)&Bs[cur][base + 16];
        }
#pragma unroll
        for (int i = 0; i < 2; ++i)
#pragma unroll
            for (int j = 0; j < 2; ++j)
                acc[i][j].v = __builtin_amdgcn_wmma_f32_16x16x32_f16(
                    false, a[i].v, false, b[j].v, (short)0, acc[i][j].v, false, false);

        wait_async();
        __syncthreads();
        cur = nxt;
    }

#pragma unroll
    for (int i = 0; i < 2; ++i)
#pragma unroll
        for (int j = 0; j < 2; ++j)
#pragma unroll
            for (int v = 0; v < 8; ++v) {
                const int row = m0 + wm * 32 + i * 16 + v + lhi * 8;
                const int col = n0 + wn * 32 + j * 16 + lmod;
                if (row < M && col < N) {
                    const float val = acc[i][j].f[v];
                    C[(size_t)row * ldc + col] = val;
                    if (Ch) Ch[(size_t)row * ldc + col] = (_Float16)val;
                }
            }
}

// ---------------------------------------------------------------------------
// dt = softplus(dt_raw + dt_bias);  dA = exp(dt * (-exp(A_log)))
// ---------------------------------------------------------------------------
__global__ void dt_kernel(const float* __restrict__ zxb, const float* __restrict__ dt_bias,
                          const float* __restrict__ A_log,
                          float* __restrict__ dtA, float* __restrict__ dAA, int total)
{
    int idx = blockIdx.x * blockDim.x + threadIdx.x;
    if (idx >= total) return;
    const int h   = idx & (NH - 1);
    const int row = idx / NH;
    const float raw = zxb[(size_t)row * DINPROJ + (DIN + CONVDIM) + h] + dt_bias[h];
    const float dt  = (raw > 20.f) ? raw : log1pf(__expf(raw));
    const float Aneg = -__expf(A_log[h]);
    dtA[idx] = dt;
    dAA[idx] = __expf(dt * Aneg);
}

// ---------------------------------------------------------------------------
// Depthwise causal conv1d (k=4) along sequence + SiLU, on the xBC slice
// ---------------------------------------------------------------------------
__global__ void dwconv_kernel(const float* __restrict__ zxb, const float* __restrict__ w,
                              const float* __restrict__ bias, float* __restrict__ xact, int total)
{
    int idx = blockIdx.x * blockDim.x + threadIdx.x;
    if (idx >= total) return;
    const int c   = idx % CONVDIM;
    const int row = idx / CONVDIM;
    const int l   = row & (SEQ - 1);
    float s = bias[c];
#pragma unroll
    for (int j = 0; j < 4; ++j) {
        const int ls = l - 3 + j;
        if (ls >= 0)
            s += w[c * 4 + j] * zxb[(size_t)(row - l + ls) * DINPROJ + DIN + c];
    }
    xact[idx] = siluf(s);
}

// ---------------------------------------------------------------------------
// Sequential SSM scan. One block per (batch, head). State (64x64) in VGPRs:
// 256 lanes x 16 floats. B/C/x staged in LDS per step; y reduced across the
// 4 lanes sharing a p-row via wave32 shfl_xor.
// ---------------------------------------------------------------------------
__global__ __launch_bounds__(256)
void scan_kernel(const float* __restrict__ xact, const float* __restrict__ dtA,
                 const float* __restrict__ dAA, const float* __restrict__ Dp,
                 float* __restrict__ Y)
{
    const int b  = blockIdx.x >> 4;
    const int hh = blockIdx.x & 15;
    const int tid = threadIdx.x;
    const int p  = tid >> 2;     // 0..63 head-dim row
    const int ng = tid & 3;      // which 16-wide n-slice

    __shared__ float xs[HEADDIM], Bsh[DSTATE], Csh[DSTATE], sc[2];

    float hstate[16];
#pragma unroll
    for (int e = 0; e < 16; ++e) hstate[e] = 0.f;
    const float Dh = Dp[hh];

    for (int t = 0; t < SEQ; ++t) {
        const int row = b * SEQ + t;
        const float* xr = xact + (size_t)row * CONVDIM;
        if (tid < 64)        xs[tid]        = xr[hh * 64 + tid];
        else if (tid < 128)  Bsh[tid - 64]  = xr[DIN + (tid - 64)];
        else if (tid < 192)  Csh[tid - 128] = xr[DIN + DSTATE + (tid - 128)];
        else if (tid == 192) { sc[0] = dAA[(size_t)row * NH + hh];
                               sc[1] = dtA[(size_t)row * NH + hh]; }
        __syncthreads();

        const float dA   = sc[0];
        const float xp   = xs[p];
        const float coef = sc[1] * xp;
        float acc = 0.f;
#pragma unroll
        for (int e = 0; e < 16; ++e) {
            const int n = ng * 16 + e;
            hstate[e] = hstate[e] * dA + coef * Bsh[n];
            acc += hstate[e] * Csh[n];
        }
        acc += __shfl_xor(acc, 1, 32);
        acc += __shfl_xor(acc, 2, 32);
        if (ng == 0)
            Y[(size_t)row * DIN + hh * 64 + p] = acc + Dh * xp;
        __syncthreads();
    }
}

// ---------------------------------------------------------------------------
// y *= silu(z); RMS-norm over DIN; write f16 A-operand for out_proj GEMM
// ---------------------------------------------------------------------------
__global__ __launch_bounds__(256)
void gate_rms_kernel(const float* __restrict__ Y, const float* __restrict__ zxb,
                     const float* __restrict__ rms_w, _Float16* __restrict__ yn)
{
    __shared__ float red[256];
    const int row = blockIdx.x, tid = threadIdx.x;
    float v[4];
    float ss = 0.f;
#pragma unroll
    for (int q = 0; q < 4; ++q) {
        const int c = tid + q * 256;
        float yv = Y[(size_t)row * DIN + c];
        const float zv = zxb[(size_t)row * DINPROJ + c];
        yv *= siluf(zv);
        v[q] = yv;
        ss += yv * yv;
    }
    red[tid] = ss;
    __syncthreads();
    for (int s = 128; s > 0; s >>= 1) {
        if (tid < s) red[tid] += red[tid + s];
        __syncthreads();
    }
    const float scale = rsqrtf(red[0] / (float)DIN + EPS_F);
#pragma unroll
    for (int q = 0; q < 4; ++q) {
        const int c = tid + q * 256;
        yn[(size_t)row * DIN + c] = (_Float16)(v[q] * scale * rms_w[c]);
    }
}

// ---------------------------------------------------------------------------
// Final conv1d as 3 accumulated WMMA GEMMs over shifted activations.
// u is (N=16, C=1024, H=512); M-rows = (n,h); K = channels; N = out-channels.
// B staged via async LDS-DMA; A is a boundary-guarded transposing gather.
// ---------------------------------------------------------------------------
__global__ __launch_bounds__(256)
void conv_gemm_wmma(const _Float16* __restrict__ U, const _Float16* __restrict__ Wc,
                    const float* __restrict__ conv_b, float* __restrict__ Cc)
{
    __shared__ __attribute__((aligned(16))) _Float16 As[128 * 40];
    __shared__ __attribute__((aligned(16))) _Float16 Bs[64 * 40];

    const int tid  = threadIdx.x;
    const int m0   = blockIdx.y * 128;     // m = n*HCONV + h ; 128 | HCONV
    const int n0   = blockIdx.x * 64;      // out-channel tile
    const int nb   = m0 / HCONV;
    const int h0   = m0 & (HCONV - 1);
    const int lane = tid & 31, wave = tid >> 5;
    const int wm = wave & 3, wn = wave >> 2;
    const int lmod = lane & 15, lhi = (lane >> 4) & 1;
    const int kr = tid >> 3;   // 0..31 channel row within K-slice
    const int mg = tid & 7;    // 0..7  sixteen-row m group
    const int brow = tid >> 2, bq = tid & 3;

    FragF acc[2][2];
    v8f zero = {};
#pragma unroll
    for (int i = 0; i < 2; ++i)
#pragma unroll
        for (int j = 0; j < 2; ++j) acc[i][j].v = zero;

    for (int tap = 0; tap < 3; ++tap) {
        const _Float16* Wt = Wc + (size_t)tap * OUT_CH * IN_CH;
        for (int kc = 0; kc < IN_CH; kc += 32) {
            // Stage B via async LDS-DMA (tiles always full: 64 | OUT_CH)
            copy16(&Bs[brow * 40 + bq * 8],
                   Wt + (size_t)(n0 + brow) * IN_CH + kc + bq * 8);

            // Stage A (transposed/shifted gather, boundary-guarded)
            const int cch = kc + kr;
            const _Float16* src = U + (size_t)nb * IN_CH * HCONV + (size_t)cch * HCONV;
            if (kc + 32 < IN_CH)
                __builtin_prefetch(src + 32 * HCONV, 0, 1);
#pragma unroll
            for (int e = 0; e < 16; ++e) {
                const int hh = h0 + mg * 16 + e + tap - 1;
                _Float16 val = (_Float16)0.f;
                if (hh >= 0 && hh < HCONV) val = src[hh];
                As[(mg * 16 + e) * 40 + kr] = val;
            }
            wait_async();
            __syncthreads();

            FragH a[2], b[2];
#pragma unroll
            for (int i = 0; i < 2; ++i) {
                const int base = (wm * 32 + i * 16 + lmod) * 40 + lhi * 8;
                a[i].q[0] = *(const u32x4*)&As[base];
                a[i].q[1] = *(const u32x4*)&As[base + 16];
            }
#pragma unroll
            for (int j = 0; j < 2; ++j) {
                const int base = (wn * 32 + j * 16 + lmod) * 40 + lhi * 8;
                b[j].q[0] = *(const u32x4*)&Bs[base];
                b[j].q[1] = *(const u32x4*)&Bs[base + 16];
            }
#pragma unroll
            for (int i = 0; i < 2; ++i)
#pragma unroll
                for (int j = 0; j < 2; ++j)
                    acc[i][j].v = __builtin_amdgcn_wmma_f32_16x16x32_f16(
                        false, a[i].v, false, b[j].v, (short)0, acc[i][j].v, false, false);
            __syncthreads();
        }
    }

    // Store to (n, oc, h) layout with conv bias added
#pragma unroll
    for (int i = 0; i < 2; ++i)
#pragma unroll
        for (int j = 0; j < 2; ++j)
#pragma unroll
            for (int v = 0; v < 8; ++v) {
                const int row = m0 + wm * 32 + i * 16 + v + lhi * 8;
                const int hh  = row & (HCONV - 1);
                const int col = n0 + wn * 32 + j * 16 + lmod;
                Cc[(size_t)nb * OUT_CH * HCONV + (size_t)col * HCONV + hh] =
                    acc[i][j].f[v] + conv_b[col];
            }
}

// ---------------------------------------------------------------------------
// BatchNorm over (N,H) per channel: pass 1 stats, pass 2 normalize + SiLU
// ---------------------------------------------------------------------------
__global__ __launch_bounds__(256)
void bn_stats_kernel(const float* __restrict__ Cc, float* __restrict__ stats)
{
    __shared__ float s1[256], s2[256];
    const int oc = blockIdx.x;
    const int tid = threadIdx.x;
    float a = 0.f, b = 0.f;
    for (int i = tid; i < BATCH * HCONV; i += 256) {
        const int n = i >> 9, hh = i & (HCONV - 1);
        const float v = Cc[(size_t)n * OUT_CH * HCONV + (size_t)oc * HCONV + hh];
        a += v; b += v * v;
    }
    s1[tid] = a; s2[tid] = b;
    __syncthreads();
    for (int s = 128; s > 0; s >>= 1) {
        if (tid < s) { s1[tid] += s1[tid + s]; s2[tid] += s2[tid + s]; }
        __syncthreads();
    }
    if (tid == 0) {
        const float inv = 1.0f / (float)(BATCH * HCONV);
        const float mean = s1[0] * inv;
        const float var  = s2[0] * inv - mean * mean;
        stats[oc]          = mean;
        stats[OUT_CH + oc] = rsqrtf(var + EPS_F);
    }
}

__global__ void bn_apply_kernel(const float* __restrict__ Cc, const float* __restrict__ stats,
                                const float* __restrict__ gamma, const float* __restrict__ beta,
                                float* __restrict__ out, int n)
{
    int idx = blockIdx.x * blockDim.x + threadIdx.x;
    if (idx >= n) return;
    const int oc = (idx >> 9) & (OUT_CH - 1);
    const float v = (Cc[idx] - stats[oc]) * stats[OUT_CH + oc] * gamma[oc] + beta[oc];
    out[idx] = siluf(v);
}

// ---------------------------------------------------------------------------
// Host-side pipeline
// ---------------------------------------------------------------------------
extern "C" void kernel_launch(void* const* d_in, const int* in_sizes, int n_in,
                              void* d_out, int out_size, void* d_ws, size_t ws_size,
                              hipStream_t stream)
{
    (void)in_sizes; (void)n_in; (void)out_size; (void)ws_size;
    const float* x        = (const float*)d_in[0];
    const float* in_proj  = (const float*)d_in[1];
    const float* conv1d_w = (const float*)d_in[2];
    const float* conv1d_b = (const float*)d_in[3];
    const float* dt_bias  = (const float*)d_in[4];
    const float* A_log    = (const float*)d_in[5];
    const float* Dp       = (const float*)d_in[6];
    const float* rms_w    = (const float*)d_in[7];
    const float* out_proj = (const float*)d_in[8];
    const float* conv_w   = (const float*)d_in[9];
    const float* conv_b   = (const float*)d_in[10];
    const float* bn_gamma = (const float*)d_in[11];
    const float* bn_beta  = (const float*)d_in[12];
    float* out = (float*)d_out;

    char* ws = (char*)d_ws;
    size_t off = 0;
    auto take = [&](size_t bytes) -> char* {
        char* p = ws + off;
        off = (off + bytes + 255) & ~(size_t)255;
        return p;
    };
    _Float16* x16   = (_Float16*)take((size_t)MROWS * DM * 2);
    _Float16* w116  = (_Float16*)take((size_t)DINPROJ * DM * 2);
    _Float16* w216  = (_Float16*)take((size_t)DM * DIN * 2);
    _Float16* wc16  = (_Float16*)take((size_t)3 * OUT_CH * IN_CH * 2);
    float*    zxb   = (float*)take((size_t)MROWS * DINPROJ * 4);
    float*    dtA   = (float*)take((size_t)MROWS * NH * 4);
    float*    dAA   = (float*)take((size_t)MROWS * NH * 4);
    float*    xact  = (float*)take((size_t)MROWS * CONVDIM * 4);
    float*    Y     = (float*)take((size_t)MROWS * DIN * 4);
    _Float16* yn16  = (_Float16*)take((size_t)MROWS * DIN * 2);
    float*    stats = (float*)take((size_t)2 * OUT_CH * 4);
    // Reuse dead regions (all producer/consumer ordering respected in-stream):
    float*    u32p = zxb;                                   // out_proj output (33.5MB of 143MB)
    _Float16* u16  = (_Float16*)(zxb + (size_t)MROWS * DM); // f16 copy right after
    float*    Cc   = Y;                                     // conv output over dead Y

    auto blocks = [](size_t n, int b) { return (unsigned)((n + (size_t)b - 1) / (size_t)b); };

    // Pack operands to f16
    cvt_f32_f16<<<blocks((size_t)MROWS * DM, 256), 256, 0, stream>>>(x, x16, MROWS * DM);
    cvt_f32_f16<<<blocks((size_t)DINPROJ * DM, 256), 256, 0, stream>>>(in_proj, w116, DINPROJ * DM);
    cvt_f32_f16<<<blocks((size_t)DM * DIN, 256), 256, 0, stream>>>(out_proj, w216, DM * DIN);
    pack_convw<<<blocks((size_t)3 * OUT_CH * IN_CH, 256), 256, 0, stream>>>(conv_w, wc16, 3 * OUT_CH * IN_CH);

    // in_proj GEMM: (16384 x 512) x (512 x 2192)
    gemm_f16_wmma<<<dim3((DINPROJ + 63) / 64, MROWS / 128), 256, 0, stream>>>(
        x16, w116, zxb, (_Float16*)nullptr, MROWS, DINPROJ, DM, DM, DM, DINPROJ);

    dt_kernel<<<blocks((size_t)MROWS * NH, 256), 256, 0, stream>>>(zxb, dt_bias, A_log, dtA, dAA, MROWS * NH);
    dwconv_kernel<<<blocks((size_t)MROWS * CONVDIM, 256), 256, 0, stream>>>(zxb, conv1d_w, conv1d_b, xact, MROWS * CONVDIM);

    // Sequential selective scan: 256 independent (b,h) recurrences
    scan_kernel<<<BATCH * NH, 256, 0, stream>>>(xact, dtA, dAA, Dp, Y);

    gate_rms_kernel<<<MROWS, 256, 0, stream>>>(Y, zxb, rms_w, yn16);

    // out_proj GEMM: (16384 x 1024) x (1024 x 512), emits f32 + f16 copies
    gemm_f16_wmma<<<dim3(DM / 64, MROWS / 128), 256, 0, stream>>>(
        yn16, w216, u32p, u16, MROWS, DM, DIN, DIN, DIN, DM);

    // conv1d as 3 accumulated GEMMs: M=8192, N=1024, K=3*1024
    conv_gemm_wmma<<<dim3(OUT_CH / 64, (BATCH * HCONV) / 128), 256, 0, stream>>>(u16, wc16, conv_b, Cc);

    bn_stats_kernel<<<OUT_CH, 256, 0, stream>>>(Cc, stats);
    bn_apply_kernel<<<blocks((size_t)BATCH * OUT_CH * HCONV, 256), 256, 0, stream>>>(
        Cc, stats, bn_gamma, bn_beta, out, BATCH * OUT_CH * HCONV);
}
